// SFNO_block_49168785604702
// MI455X (gfx1250) — matrix operand order repaired and checked by reading, not compiled, not run on hardware
//
#include <hip/hip_runtime.h>
#include <math.h>

// ---------------- problem constants ----------------
#define CC     256      // channels / filters
#define NLATD  256
#define NLOND  512
#define LM     256      // l modes
#define MMODES 257      // rfft modes
#define MPAD   272      // M padded to multiple of 16
#define KPAD   544      // 2*MPAD (re|im concatenated K for irfft GEMM)
#define NREFD  64
#define NPIX   131072   // NLON*NLAT
#define GN_EPS 1e-3f

typedef _Float16 v16h __attribute__((ext_vector_type(16)));
typedef float    v8f  __attribute__((ext_vector_type(8)));

// ---------------- batched WMMA GEMM ----------------
#define BM 128
#define BN 128
#define BK 32
#define PITCH 48   // 32 + 16 halfs: keeps fragment reads 32B-aligned, staggers banks

enum { GF_ACC = 1, GF_GELU = 2, GF_F16 = 4, GF_RES = 8, GF_BIAS = 16 };

struct GemmP {
  const _Float16* A; long lda, sA;     // row-major M x K, per-batch stride
  const _Float16* B; long ldb, sB;     // row-major K x N, per-batch stride
  void*           D; long dB, dR, dC;  // affine element strides (batch,row,col)
  const _Float16* Res; long rB, rR, rC;
  const float*    bias;                // per output column
  int M, N, K, flags;
  float alpha;
};

__device__ __forceinline__ float gelu_exact(float x) {
  return 0.5f * x * (1.0f + erff(x * 0.70710678118654752f));
}

union Frag { uint4 q[2]; v16h v; };

template <bool GUARD>
__device__ __forceinline__ void gemm_epilogue(const GemmP& p, int b, long row0, long col0,
                                              int wr, int wc, int lane, v8f (&acc)[2][4]) {
  const int l16 = lane & 15;
  const int mh  = (lane >> 4) * 8;
#pragma unroll
  for (int i = 0; i < 2; ++i)
#pragma unroll
    for (int j = 0; j < 4; ++j) {
      const long c = col0 + wc * 64 + j * 16 + l16;
      if (GUARD && c >= p.N) continue;
      const long rbase = row0 + wr * 32 + i * 16 + mh;
#pragma unroll
      for (int v = 0; v < 8; ++v) {
        const long r = rbase + v;
        if (GUARD && r >= p.M) continue;
        float val = acc[i][j][v] * p.alpha;
        const long off = (long)b * p.dB + r * p.dR + c * p.dC;
        if (p.flags & GF_ACC) {
          ((float*)p.D)[off] += val;
        } else {
          if (p.flags & GF_BIAS) val += p.bias[c];
          if (p.flags & GF_GELU) val = gelu_exact(val);
          if (p.flags & GF_RES)  val += (float)p.Res[(long)b * p.rB + r * p.rR + c * p.rC];
          if (p.flags & GF_F16) ((_Float16*)p.D)[off] = (_Float16)val;
          else                  ((float*)p.D)[off] = val;
        }
      }
    }
}

__global__ __launch_bounds__(256) void gemm_wmma(GemmP p) {
  __shared__ _Float16 As[2][BM][PITCH];   // [buf][row][k]
  __shared__ _Float16 Bs[2][BN][PITCH];   // [buf][col][k]  (transposed)
  const int t    = threadIdx.x;
  const int lane = t & 31;
  const int wid  = t >> 5;
  const int wr   = wid & 3;            // 4 row-waves x 32 rows
  const int wc   = wid >> 2;           // 2 col-waves x 64 cols
  const int b    = blockIdx.z;
  const long row0 = (long)blockIdx.y * BM;
  const long col0 = (long)blockIdx.x * BN;
  const _Float16* Ab = p.A + (long)b * p.sA;
  const _Float16* Bb = p.B + (long)b * p.sB;
  const bool fullM = (row0 + BM <= p.M);
  const bool fullN = (col0 + BN <= p.N);

  // staging coordinates
  const int ar  = t >> 2;              // A: rows 0..63 (+64 second chunk)
  const int ak  = (t & 3) * 8;         // A: k chunk of 8 halfs
  const int bkr = t >> 4;              // B: k rows 0..15 (+16 second chunk)
  const int bc  = (t & 15) * 8;        // B: col chunk of 8 halfs

  const _Float16* aP = Ab + (row0 + ar) * p.lda + ak;
  const _Float16* bP = Bb + (long)bkr * p.ldb + col0 + bc;
  const long aRow2 = 64 * p.lda;
  const long bRow2 = 16 * p.ldb;
  const long bStep = (long)BK * p.ldb;

  // per-buffer LDS byte addresses for async A staging
  unsigned aDst0[2], aDst1[2];
#pragma unroll
  for (int u = 0; u < 2; ++u) {
    aDst0[u] = (unsigned)(uintptr_t)&As[u][ar][ak];
    aDst1[u] = (unsigned)(uintptr_t)&As[u][ar + 64][ak];
  }

  auto stage = [&](int buf, const _Float16* ap, const _Float16* bp) {
    // ---- A tile ----
    if (fullM) {
      // CDNA5 async DMA into LDS (ASYNCcnt-tracked), ISA 15.18.3 op 98
      unsigned long long g0 = (unsigned long long)(uintptr_t)ap;
      unsigned long long g1 = (unsigned long long)(uintptr_t)(ap + aRow2);
      asm volatile("global_load_async_to_lds_b128 %0, %1, off"
                   :: "v"(aDst0[buf]), "v"(g0) : "memory");
      asm volatile("global_load_async_to_lds_b128 %0, %1, off"
                   :: "v"(aDst1[buf]), "v"(g1) : "memory");
    } else {
#pragma unroll
      for (int it = 0; it < 2; ++it) {
        const long gr = row0 + ar + it * 64;
        uint4 v = make_uint4(0u, 0u, 0u, 0u);
        if (gr < p.M) v = *(const uint4*)(ap + it * aRow2);
        *(uint4*)&As[buf][ar + it * 64][ak] = v;
      }
    }
    // ---- B tile: transpose into LDS so fragments are K-contiguous per column ----
#pragma unroll
    for (int it = 0; it < 2; ++it) {
      const _Float16* src = bp + it * bRow2;
      const int kk = bkr + it * 16;
      _Float16 tmp[8];
      if (fullN || (col0 + bc + 8 <= p.N)) {
        *(uint4*)tmp = *(const uint4*)src;
      } else {
#pragma unroll
        for (int i = 0; i < 8; ++i) tmp[i] = (col0 + bc + i < p.N) ? src[i] : (_Float16)0.0f;
      }
#pragma unroll
      for (int i = 0; i < 8; ++i) Bs[buf][bc + i][kk] = tmp[i];
    }
  };

  const v8f vzero = {0.f,0.f,0.f,0.f,0.f,0.f,0.f,0.f};
  v8f acc[2][4];
#pragma unroll
  for (int i = 0; i < 2; ++i)
#pragma unroll
    for (int j = 0; j < 4; ++j) acc[i][j] = vzero;

  const int nK = p.K / BK;            // all K in this pipeline are multiples of 32

  stage(0, aP, bP);
  aP += BK; bP += bStep;
  asm volatile("s_wait_asynccnt 0x0" ::: "memory");
  __syncthreads();

  const int g   = lane >> 4;          // half-wave
  const int l16 = lane & 15;

  for (int kt = 0; kt < nK; ++kt) {
    const int cur = kt & 1;
    if (kt + 1 < nK) {
      stage(cur ^ 1, aP, bP);         // overlap next tile DMA with this tile's WMMA
      aP += BK; bP += bStep;
      __builtin_prefetch((const void*)aP, 0, 1);   // global_prefetch_b8 for kt+2
      __builtin_prefetch((const void*)bP, 0, 1);
    }
    // ---- fragments per CDNA5 WMMA VGPR layouts ----
    Frag af[2], bf[4];
#pragma unroll
    for (int i = 0; i < 2; ++i) {
      // 16-bit A 16x32: lanes 0-15 hold K {0..7,16..23}; lanes 16-31 hold K {8..15,24..31}
      const _Float16* apf = &As[cur][wr * 32 + i * 16 + l16][0];
      af[i].q[0] = *(const uint4*)(apf + g * 8);
      af[i].q[1] = *(const uint4*)(apf + 16 + g * 8);
    }
#pragma unroll
    for (int j = 0; j < 4; ++j) {
      // 16-bit B 32x16: lane = column; lanes 0-15 hold K {0..7,16..23}; lanes 16-31 hold K {8..15,24..31}
      const _Float16* bpf = &Bs[cur][wc * 64 + j * 16 + l16][0];
      bf[j].q[0] = *(const uint4*)(bpf + g * 8);
      bf[j].q[1] = *(const uint4*)(bpf + 16 + g * 8);
    }
#pragma unroll
    for (int i = 0; i < 2; ++i)
#pragma unroll
      for (int j = 0; j < 4; ++j)
        acc[i][j] = __builtin_amdgcn_wmma_f32_16x16x32_f16(
            false, af[i].v, false, bf[j].v, (short)0, acc[i][j], false, false);
    asm volatile("s_wait_asynccnt 0x0" ::: "memory");
    __syncthreads();
  }

  if (fullM && fullN) gemm_epilogue<false>(p, b, row0, col0, wr, wc, lane, acc);
  else                gemm_epilogue<true >(p, b, row0, col0, wr, wc, lane, acc);
}

// ---------------- elementwise / prep kernels ----------------
__global__ void k_convert_x(const float* __restrict__ x, _Float16* x16, _Float16* xt16, long n) {
  long i = (long)blockIdx.x * 256 + threadIdx.x;
  if (i >= n) return;
  float v = x[i];
  x16[i] = (_Float16)v;
  long c = i >> 17, rem = i & 131071, j = rem >> 9, nn = rem & 511;
  xt16[(nn * NLATD + j) * CC + c] = (_Float16)v;
}

__global__ void k_build_dft(_Float16* B1, long n) {  // (NLON x KPAD): cols 0..271 cos, 272.. -sin
  long i = (long)blockIdx.x * 256 + threadIdx.x;
  if (i >= n) return;
  int nn = (int)(i / KPAD), k = (int)(i % KPAD);
  int m = (k < MPAD) ? k : k - MPAD;
  int ph = (nn * m) & 511;                       // exact phase reduction
  float th = 6.283185307179586f * (float)ph / 512.0f;
  B1[i] = (_Float16)((k < MPAD) ? cosf(th) : -sinf(th));
}

__global__ void k_build_idft(_Float16* ID, long n) { // (KPAD x NLON) irfft weights
  long i = (long)blockIdx.x * 256 + threadIdx.x;
  if (i >= n) return;
  int k = (int)(i / NLOND), nn = (int)(i % NLOND);
  int m = (k < MPAD) ? k : k - MPAD;
  float v = 0.0f;
  if (m < MMODES) {
    float w = ((m == 0) || (m == 256)) ? 1.0f : 2.0f;
    w *= (1.0f / 512.0f);
    int ph = (nn * m) & 511;
    float th = 6.283185307179586f * (float)ph / 512.0f;
    v = (k < MPAD) ? w * cosf(th) : -w * sinf(th);
  }
  ID[i] = (_Float16)v;
}

__global__ void k_build_leg(const float* __restrict__ legf, const float* __restrict__ wq,
                            _Float16* lw16, _Float16* leg16, long n) {
  long i = (long)blockIdx.x * 256 + threadIdx.x;
  if (i >= n) return;
  int m = (int)(i >> 16), r = (int)(i & 65535), l = r >> 8, j = r & 255;
  float v = legf[((long)l * MMODES + m) * NLATD + j];
  lw16[(long)m * 65536 + (long)j * 256 + l]  = (_Float16)(v * wq[j]);  // [m][j][l] (KxN)
  leg16[(long)m * 65536 + (long)l * 256 + j] = (_Float16)v;            // [m][l][j] (KxN)
}

__global__ void k_build_filters(const float* __restrict__ wre, const float* __restrict__ wim,
                                _Float16* Wr16, _Float16* Wi16, long n) {
  long i = (long)blockIdx.x * 256 + threadIdx.x;
  if (i >= n) return;
  int l = (int)(i >> 16), fc = (int)(i & 65535);
  float pos = (float)l * (63.0f / 255.0f);
  int i0 = (int)pos; if (i0 > 62) i0 = 62;
  float fr = pos - (float)i0;
  long base = (long)fc * NREFD;
  Wr16[(long)l * 65536 + fc] = (_Float16)(wre[base + i0] * (1.0f - fr) + wre[base + i0 + 1] * fr);
  Wi16[(long)l * 65536 + fc] = (_Float16)(wim[base + i0] * (1.0f - fr) + wim[base + i0 + 1] * fr);
}

__global__ void k_f32_to_f16(const float* __restrict__ s, _Float16* d, long n) {
  long i = (long)blockIdx.x * 256 + threadIdx.x;
  if (i < n) d[i] = (_Float16)s[i];
}

__global__ void k_fill_zero_h(_Float16* d, long n) {
  long i = (long)blockIdx.x * 256 + threadIdx.x;
  if (i < n) d[i] = (_Float16)0.0f;
}

__global__ void k_pack_xs(const float* __restrict__ xs32, _Float16* xs16, long n) {
  long i = (long)blockIdx.x * 256 + threadIdx.x;
  if (i >= n) return;
  int l = (int)(i & 255);                       // layout [m][f][l]
  float s = sqrtf(1.0f + (float)l) * (1.0f / 256.0f);   // lfac / C
  xs16[i] = (_Float16)(xs32[i] * s);
}

__global__ void k_stats(const _Float16* __restrict__ y2, float* mu, float* rs) {
  int c = blockIdx.x, t = threadIdx.x;
  float s = 0.f, ss = 0.f;
  for (int r = t; r < NPIX; r += 256) {
    float v = (float)y2[(long)r * CC + c];
    s += v; ss += v * v;
  }
  __shared__ float sh[256], sh2[256];
  sh[t] = s; sh2[t] = ss; __syncthreads();
  for (int d = 128; d > 0; d >>= 1) {
    if (t < d) { sh[t] += sh[t + d]; sh2[t] += sh2[t + d]; }
    __syncthreads();
  }
  if (t == 0) {
    float m = sh[0] / (float)NPIX;
    float var = sh2[0] / (float)NPIX - m * m;
    mu[c] = m; rs[c] = rsqrtf(var + GN_EPS);
  }
}

__global__ void k_finalize(const _Float16* __restrict__ y2, const float* __restrict__ x,
                           const float* __restrict__ mu, const float* __restrict__ rs,
                           const float* __restrict__ gm, const float* __restrict__ bt,
                           float* out, long n) {
  long i = (long)blockIdx.x * 256 + threadIdx.x;
  if (i >= n) return;
  long c = i >> 17, rem = i & 131071, j = rem >> 9, nn = rem & 511;
  float v = (float)y2[(nn * NLATD + j) * CC + c];
  out[i] = (v - mu[c]) * rs[c] * gm[c] + bt[c] + x[i];
}

// ---------------- host orchestration ----------------
static inline void launch_gemm(hipStream_t s, GemmP p, int batches) {
  dim3 grid((unsigned)((p.N + BN - 1) / BN), (unsigned)((p.M + BM - 1) / BM), (unsigned)batches);
  gemm_wmma<<<grid, dim3(256), 0, s>>>(p);
}
#define EW(kern, n, ...) kern<<<(unsigned)(((n) + 255) / 256), 256, 0, stream>>>(__VA_ARGS__, (long)(n))

extern "C" void kernel_launch(void* const* d_in, const int* in_sizes, int n_in,
                              void* d_out, int out_size, void* d_ws, size_t ws_size,
                              hipStream_t stream) {
  (void)in_sizes; (void)n_in; (void)out_size; (void)ws_size;
  const float* x     = (const float*)d_in[0];
  const float* legf  = (const float*)d_in[1];
  const float* wq    = (const float*)d_in[2];
  const float* wre   = (const float*)d_in[3];
  const float* wim   = (const float*)d_in[4];
  const float* m1w1  = (const float*)d_in[5];
  const float* m1b1  = (const float*)d_in[6];
  const float* m1w2  = (const float*)d_in[7];
  const float* m1b2  = (const float*)d_in[8];
  const float* m2w1  = (const float*)d_in[9];
  const float* m2b1  = (const float*)d_in[10];
  const float* m2w2  = (const float*)d_in[11];
  const float* m2b2  = (const float*)d_in[12];
  const float* gmm   = (const float*)d_in[13];
  const float* bet   = (const float*)d_in[14];
  float* out = (float*)d_out;

  // bump allocator over d_ws (256B aligned regions)
  char* W = (char*)d_ws;
  size_t off = 0;
  auto take = [&](size_t bytes) -> char* {
    char* p = W + off;
    off = (off + bytes + 255) & ~(size_t)255;
    return p;
  };
  _Float16* x16   = (_Float16*)take(2ul * CC * NLATD * NLOND);         // f16 copy of x (C,NLAT,NLON)
  _Float16* xt16  = (_Float16*)take(2ul * NPIX * CC);                  // x transposed (n,j,c)
  _Float16* B1    = (_Float16*)take(2ul * NLOND * KPAD);               // rDFT matrix
  _Float16* IDF   = (_Float16*)take(2ul * KPAD * NLOND);               // irDFT matrix
  _Float16* lw16  = (_Float16*)take(2ul * MMODES * NLATD * LM);        // [m][j][l] leg*wq
  _Float16* leg16 = (_Float16*)take(2ul * MMODES * LM * NLATD);        // [m][l][j]
  _Float16* Wr16  = (_Float16*)take(2ul * LM * CC * CC);               // [l][f][c]
  _Float16* Wi16  = (_Float16*)take(2ul * LM * CC * CC);
  _Float16* Fall  = (_Float16*)take(2ul * KPAD * CC * NLATD);          // [k][c][j]  (re|im spectra)
  _Float16* hr16  = (_Float16*)take(2ul * LM * CC * MPAD);             // [l][c][m]
  _Float16* hi16  = (_Float16*)take(2ul * LM * CC * MPAD);
  float*    xs32r = (float*)take(4ul * MPAD * CC * LM);                // [m][f][l]
  float*    xs32i = (float*)take(4ul * MPAD * CC * LM);
  _Float16* xs16r = (_Float16*)take(2ul * MPAD * CC * LM);
  _Float16* xs16i = (_Float16*)take(2ul * MPAD * CC * LM);
  _Float16* A5    = (_Float16*)take(2ul * (size_t)CC * NLATD * KPAD);  // [(f,j)][k]
  _Float16* mw1   = (_Float16*)take(2ul * CC * CC);
  _Float16* mw2   = (_Float16*)take(2ul * CC * CC);
  _Float16* mw3   = (_Float16*)take(2ul * CC * CC);
  _Float16* mw4   = (_Float16*)take(2ul * CC * CC);
  float*    mu    = (float*)take(4ul * CC);
  float*    rsg   = (float*)take(4ul * CC);
  // aliased regions (lifetimes do not overlap):
  _Float16* g16   = hr16;                // 67MB into 71MB (hr16+hi16), written at stage 5
  _Float16* h1    = x16;                 // x16 dead after stage 1
  _Float16* added = (_Float16*)xs32r;    // dead after pack
  _Float16* h2    = (_Float16*)xs32i;    // dead after pack
  _Float16* y2    = Fall;                // dead after stage 2

  // ---- prep ----
  EW(k_convert_x, (long)CC * NLATD * NLOND, x, x16, xt16);
  EW(k_build_dft, (long)NLOND * KPAD, B1);
  EW(k_build_idft, (long)KPAD * NLOND, IDF);
  EW(k_build_leg, (long)MMODES * 65536, legf, wq, lw16, leg16);
  EW(k_build_filters, (long)LM * 65536, wre, wim, Wr16, Wi16);
  EW(k_f32_to_f16, (long)CC * CC, m1w1, mw1);
  EW(k_f32_to_f16, (long)CC * CC, m1w2, mw2);
  EW(k_f32_to_f16, (long)CC * CC, m2w1, mw3);
  EW(k_f32_to_f16, (long)CC * CC, m2w2, mw4);
  EW(k_fill_zero_h, (long)CC * NLATD * KPAD, A5);   // padded K columns must be exactly zero

  GemmP p{};
  // ---- stage 1: rfft as DFT: per channel c, (NLAT x NLON) @ (NLON x KPAD) -> Fall[k][c][j]
  p = GemmP{x16, NLOND, (long)NLATD * NLOND, B1, KPAD, 0,
            Fall, 256, 1, 65536, nullptr, 0, 0, 0, nullptr,
            NLATD, KPAD, NLOND, GF_F16, 1.0f};
  launch_gemm(stream, p, CC);
  // ---- stage 2: forward Legendre, per m: (C x NLAT) @ (NLAT x L) -> harms[l][c][m]
  p = GemmP{Fall, 256, 65536, lw16, 256, 65536,
            hr16, 1, MPAD, (long)CC * MPAD, nullptr, 0, 0, 0, nullptr,
            CC, LM, NLATD, GF_F16, 1.0f};
  launch_gemm(stream, p, MMODES);
  p.A = Fall + (long)MPAD * CC * NLATD; p.D = hi16;
  launch_gemm(stream, p, MMODES);
  // ---- stage 3: complex spectral filter, per l: (F x C) @ (C x MPAD), f32 accumulate
  p = GemmP{Wr16, 256, 65536, hr16, MPAD, (long)CC * MPAD,
            xs32r, 1, 256, 65536, nullptr, 0, 0, 0, nullptr,
            CC, MPAD, CC, 0, 1.0f};
  launch_gemm(stream, p, LM);                                   // xs_r  = Wr*hr
  p.A = Wi16; p.B = hi16; p.flags = GF_ACC; p.alpha = -1.0f;
  launch_gemm(stream, p, LM);                                   // xs_r -= Wi*hi
  p.A = Wr16; p.B = hi16; p.D = xs32i; p.flags = 0; p.alpha = 1.0f;
  launch_gemm(stream, p, LM);                                   // xs_i  = Wr*hi
  p.A = Wi16; p.B = hr16; p.flags = GF_ACC;
  launch_gemm(stream, p, LM);                                   // xs_i += Wi*hr
  EW(k_pack_xs, (long)MPAD * CC * LM, xs32r, xs16r);            // * sqrt(1+l)/C, ->f16
  EW(k_pack_xs, (long)MPAD * CC * LM, xs32i, xs16i);
  // ---- stage 4: inverse Legendre, per m: (F x L) @ (L x NLAT) -> A5[(f,j)][m | 272+m]
  p = GemmP{xs16r, 256, 65536, leg16, 256, 65536,
            A5, 1, (long)NLATD * KPAD, KPAD, nullptr, 0, 0, 0, nullptr,
            CC, NLATD, LM, GF_F16, 1.0f};
  launch_gemm(stream, p, MMODES);
  p.A = xs16i; p.D = A5 + MPAD;
  launch_gemm(stream, p, MMODES);
  // ---- stage 5: irfft as DFT + GELU, per f: (NLAT x KPAD) @ (KPAD x NLON) -> g16[n][j][f]
  p = GemmP{A5, KPAD, (long)NLATD * KPAD, IDF, NLOND, 0,
            g16, 1, 256, 65536, nullptr, 0, 0, 0, nullptr,
            NLATD, NLOND, KPAD, GF_F16 | GF_GELU, 1.0f};
  launch_gemm(stream, p, CC);
  // ---- stage 6: MLP chain on (NPIX x C) rows ----
  p = GemmP{xt16, 256, 0, mw1, 256, 0,
            h1, 0, 256, 1, nullptr, 0, 0, 0, m1b1,
            NPIX, CC, CC, GF_F16 | GF_GELU | GF_BIAS, 1.0f};
  launch_gemm(stream, p, 1);                                    // h1 = gelu(x1@W1+b1)
  p = GemmP{h1, 256, 0, mw2, 256, 0,
            added, 0, 256, 1, g16, 0, 256, 1, m1b2,
            NPIX, CC, CC, GF_F16 | GF_GELU | GF_BIAS | GF_RES, 1.0f};
  launch_gemm(stream, p, 1);                                    // added = gelu(h1@W2+b2) + g
  p = GemmP{added, 256, 0, mw3, 256, 0,
            h2, 0, 256, 1, nullptr, 0, 0, 0, m2b1,
            NPIX, CC, CC, GF_F16 | GF_GELU | GF_BIAS, 1.0f};
  launch_gemm(stream, p, 1);
  p = GemmP{h2, 256, 0, mw4, 256, 0,
            y2, 0, 256, 1, nullptr, 0, 0, 0, m2b2,
            NPIX, CC, CC, GF_F16 | GF_GELU | GF_BIAS, 1.0f};
  launch_gemm(stream, p, 1);
  // ---- instance norm + residual ----
  k_stats<<<dim3(CC), dim3(256), 0, stream>>>(y2, mu, rsg);
  EW(k_finalize, (long)CC * NLATD * NLOND, y2, x, mu, rsg, gmm, bet, out);
}
